// WDLSTM_23983097381328
// MI455X (gfx1250) — matrix operand order repaired and checked by reading, not compiled, not run on hardware
//
#include <hip/hip_runtime.h>
#include <hip/hip_bf16.h>

// ---------------------------------------------------------------------------
// WDLSTM forward on gfx1250 (MI455X): bf16 WMMA GEMMs + fused LSTM cell scan.
// Split-K dual-chain GEMM core: two independent accumulation chains per wave
// so fragment registers cannot be aliased and loads stay in flight across
// WMMAs (descending s_wait_loadcnt ladder instead of drains to zero).
// ---------------------------------------------------------------------------

typedef __attribute__((ext_vector_type(16))) __bf16 v16bf;
typedef __attribute__((ext_vector_type(8)))  float  v8f;

static constexpr int T  = 512;
static constexpr int Bz = 64;
static constexpr int I  = 1024;
static constexpr int H  = 1024;
static constexpr int G  = 4 * H;        // 4096 gate columns
static constexpr int M1 = T * Bz;       // 32768 rows of big GEMM
static constexpr int BH = Bz * H;       // 65536

union Frag16 {                          // 16 bf16 = 32B = two 128-bit loads
    v16bf v;
    uint4 q[2];
};

__device__ __forceinline__ __bf16 f2bf(float f) {
    unsigned u = __float_as_uint(f);
    unsigned r = u + 0x7FFFu + ((u >> 16) & 1u);   // round-to-nearest-even
    unsigned short h = (unsigned short)(r >> 16);
    __bf16 b;
    __builtin_memcpy(&b, &h, 2);
    return b;
}

__device__ __forceinline__ float sigmoidf_(float x) {
    return 1.0f / (1.0f + __expf(-x));
}

#define WMMA_BF16(A, B, C) \
    __builtin_amdgcn_wmma_f32_16x16x32_bf16(false, (A), false, (B), (short)0, (C), false, false)

// ---------------------------------------------------------------------------
// Dual-chain 16x64-strip GEMM core: D(16x64) += A(16xK) * B(Kx64)^T
// Chain 1 covers K[0, K/2), chain 2 covers K[K/2, K); summed at the end.
//   arow: lane-adjusted pointer into A (row m0+l15, + hi?16:0)
//   brow: lane-adjusted pointer into B (row n0+l15, + hi?16:0)
// K and LD are compile-time so all global_load_b128 keep immediate offsets.
// ---------------------------------------------------------------------------
template <int K, int LD>
__device__ __forceinline__ void gemm_core(const __bf16* __restrict__ arow,
                                          const __bf16* __restrict__ brow,
                                          v8f& acc0, v8f& acc1, v8f& acc2, v8f& acc3) {
    constexpr int KH = K / 2;
    v8f e0 = {}, e1 = {}, e2 = {}, e3 = {};

#pragma unroll
    for (int kb = 0; kb < KH; kb += 32) {
        Frag16 a1, a2, b10, b20, b11, b21, b12, b22, b13, b23;
        // Issue every load of this iteration before any WMMA consumes one.
        a1.q[0]  = *(const uint4*)(arow + kb);
        a1.q[1]  = *(const uint4*)(arow + kb + 8);
        a2.q[0]  = *(const uint4*)(arow + KH + kb);
        a2.q[1]  = *(const uint4*)(arow + KH + kb + 8);
        b10.q[0] = *(const uint4*)(brow + 0 * 16 * LD + kb);
        b10.q[1] = *(const uint4*)(brow + 0 * 16 * LD + kb + 8);
        b20.q[0] = *(const uint4*)(brow + 0 * 16 * LD + KH + kb);
        b20.q[1] = *(const uint4*)(brow + 0 * 16 * LD + KH + kb + 8);
        b11.q[0] = *(const uint4*)(brow + 1 * 16 * LD + kb);
        b11.q[1] = *(const uint4*)(brow + 1 * 16 * LD + kb + 8);
        b21.q[0] = *(const uint4*)(brow + 1 * 16 * LD + KH + kb);
        b21.q[1] = *(const uint4*)(brow + 1 * 16 * LD + KH + kb + 8);
        b12.q[0] = *(const uint4*)(brow + 2 * 16 * LD + kb);
        b12.q[1] = *(const uint4*)(brow + 2 * 16 * LD + kb + 8);
        b22.q[0] = *(const uint4*)(brow + 2 * 16 * LD + KH + kb);
        b22.q[1] = *(const uint4*)(brow + 2 * 16 * LD + KH + kb + 8);
        b13.q[0] = *(const uint4*)(brow + 3 * 16 * LD + kb);
        b13.q[1] = *(const uint4*)(brow + 3 * 16 * LD + kb + 8);
        b23.q[0] = *(const uint4*)(brow + 3 * 16 * LD + KH + kb);
        b23.q[1] = *(const uint4*)(brow + 3 * 16 * LD + KH + kb + 8);

        acc0 = WMMA_BF16(a1.v, b10.v, acc0);
        e0   = WMMA_BF16(a2.v, b20.v, e0);
        acc1 = WMMA_BF16(a1.v, b11.v, acc1);
        e1   = WMMA_BF16(a2.v, b21.v, e1);
        acc2 = WMMA_BF16(a1.v, b12.v, acc2);
        e2   = WMMA_BF16(a2.v, b22.v, e2);
        acc3 = WMMA_BF16(a1.v, b13.v, acc3);
        e3   = WMMA_BF16(a2.v, b23.v, e3);
    }
    acc0 = acc0 + e0;
    acc1 = acc1 + e1;
    acc2 = acc2 + e2;
    acc3 = acc3 + e3;
}

// ---------------------------------------------------------------------------
// Prep: convert weights to bf16 (DropConnect folded into w_hh), sum biases.
// ---------------------------------------------------------------------------
__global__ void prep_weights(const float* __restrict__ w_ih,
                             const float* __restrict__ w_hh,
                             const float* __restrict__ hh_mask,
                             const float* __restrict__ b_ih,
                             const float* __restrict__ b_hh,
                             __bf16* __restrict__ wb_ih,
                             __bf16* __restrict__ wb_hh,
                             float* __restrict__ bias) {
    int idx = blockIdx.x * blockDim.x + threadIdx.x;
    if (idx < G * I) {
        wb_ih[idx] = f2bf(w_ih[idx]);
        wb_hh[idx] = f2bf(w_hh[idx] * hh_mask[idx]);
    }
    if (idx < G) bias[idx] = b_ih[idx] + b_hh[idx];
}

__global__ void conv_x(const float* __restrict__ x, __bf16* __restrict__ xb, int n) {
    int idx = blockIdx.x * blockDim.x + threadIdx.x;
    if (idx < n) xb[idx] = f2bf(x[idx]);
}

__global__ void init_state(float* __restrict__ c, __bf16* __restrict__ hb) {
    int idx = blockIdx.x * blockDim.x + threadIdx.x;
    if (idx < BH) {
        c[idx] = 0.0f;
        unsigned short z = 0;
        __builtin_memcpy(&hb[idx], &z, 2);
    }
}

// ---------------------------------------------------------------------------
// Big GEMM: gx[m,n] = sum_k xb[m,k] * wb_ih[n,k] + bias[n]
// One wave per 16(M) x 64(N) strip; A fragments reused across 4 B tiles.
// ---------------------------------------------------------------------------
__global__ void gx_gemm(const __bf16* __restrict__ xb,    // (M1, I) row-major
                        const __bf16* __restrict__ wb,    // (G,  I) row-major
                        const float*  __restrict__ bias,  // (G)
                        float* __restrict__ gx) {         // (M1, G) row-major
    const int MT = M1 / 16;                               // 2048 row tiles
    int wave = (blockIdx.x * blockDim.x + threadIdx.x) >> 5;
    int lane = threadIdx.x & 31;
    int l15  = lane & 15;
    bool hi  = lane >= 16;

    int m0 = (wave % MT) * 16;
    int n0 = (wave / MT) * 64;

    v8f acc0 = {}, acc1 = {}, acc2 = {}, acc3 = {};

    // A: lane holds row m0+l15; lanes 0-15 K 0..15, lanes 16-31 K 16..31
    const __bf16* arow = xb + (size_t)(m0 + l15) * I + (hi ? 16 : 0);
    // B: lane holds column n0+l15; lanes 0-15 K 0..15, lanes 16-31 K 16..31
    const __bf16* brow = wb + (size_t)(n0 + l15) * I + (hi ? 16 : 0);

    gemm_core<I, I>(arow, brow, acc0, acc1, acc2, acc3);

    int hi8 = hi ? 8 : 0;
#pragma unroll
    for (int j = 0; j < 4; ++j) {
        const v8f& acc = (j == 0) ? acc0 : (j == 1) ? acc1 : (j == 2) ? acc2 : acc3;
        int n = n0 + j * 16 + l15;
        float bn = bias[n];
        float* o = gx + (size_t)(m0 + hi8) * G + n;
#pragma unroll
        for (int v = 0; v < 8; ++v) o[(size_t)v * G] = acc[v] + bn;
    }
}

// ---------------------------------------------------------------------------
// Per-step GEMM: gates[b,n] = gx_t[b,n] + sum_k hb[b,k] * wb_hh[n,k]
// Accumulator seeded with the gx tile (C/D layout), so no extra add pass.
// Also prefetches next timestep's gx tile toward L2 (HBM-streamed operand).
// ---------------------------------------------------------------------------
__global__ void step_gemm(const __bf16* __restrict__ hb,    // (Bz, H)
                          const __bf16* __restrict__ wbh,   // (G, H)
                          const float*  __restrict__ gxt,   // (Bz, G) this t
                          float* __restrict__ gates) {      // (Bz, G)
    const int MT = Bz / 16;                                 // 4 row tiles
    int wave = (blockIdx.x * blockDim.x + threadIdx.x) >> 5;
    int lane = threadIdx.x & 31;
    int l15  = lane & 15;
    bool hi  = lane >= 16;
    int hi8  = hi ? 8 : 0;

    int m0 = (wave % MT) * 16;
    int n0 = (wave / MT) * 64;

    // Seed accumulators with the gx tile (C/D VGPR layout).
    v8f acc0, acc1, acc2, acc3;
#pragma unroll
    for (int j = 0; j < 4; ++j) {
        v8f* acc = (j == 0) ? &acc0 : (j == 1) ? &acc1 : (j == 2) ? &acc2 : &acc3;
        const float* g = gxt + (size_t)(m0 + hi8) * G + n0 + j * 16 + l15;
#pragma unroll
        for (int v = 0; v < 8; ++v) (*acc)[v] = g[(size_t)v * G];
    }

    // Prefetch next timestep's gx tile into cache (address stays inside ws:
    // at t=T-1 it lands in the xb region). Streamed from HBM otherwise.
    {
        const float* nxt = gxt + (size_t)Bz * G + (size_t)(m0 + hi8) * G + n0 + l15;
#pragma unroll
        for (int v = 0; v < 8; ++v) __builtin_prefetch(nxt + (size_t)v * G, 0, 1);
    }

    const __bf16* arow = hb  + (size_t)(m0 + l15) * H + (hi ? 16 : 0);
    const __bf16* brow = wbh + (size_t)(n0 + l15) * H + (hi ? 16 : 0);

    gemm_core<H, H>(arow, brow, acc0, acc1, acc2, acc3);

#pragma unroll
    for (int j = 0; j < 4; ++j) {
        const v8f& acc = (j == 0) ? acc0 : (j == 1) ? acc1 : (j == 2) ? acc2 : acc3;
        float* o = gates + (size_t)(m0 + hi8) * G + n0 + j * 16 + l15;
#pragma unroll
        for (int v = 0; v < 8; ++v) o[(size_t)v * G] = acc[v];
    }
}

// ---------------------------------------------------------------------------
// LSTM cell: gate activations, c/h update, bf16 h for next step, outputs.
// Gate order (PyTorch): i, f, g, o as quarters of the 4H axis.
// ---------------------------------------------------------------------------
__global__ void cell_update(const float* __restrict__ gates,  // (Bz, G)
                            float* __restrict__ c,            // (Bz, H)
                            __bf16* __restrict__ hb,          // (Bz, H)
                            float* __restrict__ out_t,        // (Bz, H)
                            float* __restrict__ hn,
                            float* __restrict__ cn,
                            int last) {
    int idx = blockIdx.x * blockDim.x + threadIdx.x;
    if (idx >= BH) return;
    int b = idx >> 10;
    int j = idx & (H - 1);
    const float* gr = gates + (size_t)b * G;
    float ig = sigmoidf_(gr[j]);
    float fg = sigmoidf_(gr[H + j]);
    float gg = tanhf(gr[2 * H + j]);
    float og = sigmoidf_(gr[3 * H + j]);
    float cnew = fg * c[idx] + ig * gg;
    float hnew = og * tanhf(cnew);
    c[idx]    = cnew;
    hb[idx]   = f2bf(hnew);
    out_t[idx] = hnew;
    if (last) { hn[idx] = hnew; cn[idx] = cnew; }
}

// ---------------------------------------------------------------------------
// Host-side launcher
// ---------------------------------------------------------------------------
extern "C" void kernel_launch(void* const* d_in, const int* in_sizes, int n_in,
                              void* d_out, int out_size, void* d_ws, size_t ws_size,
                              hipStream_t stream) {
    const float* x       = (const float*)d_in[0];
    const float* w_ih    = (const float*)d_in[1];
    const float* w_hh    = (const float*)d_in[2];
    const float* b_ih    = (const float*)d_in[3];
    const float* b_hh    = (const float*)d_in[4];
    const float* hh_mask = (const float*)d_in[5];

    float* out = (float*)d_out;              // (T, Bz, H)
    float* hn  = out + (size_t)T * BH;       // (Bz, H)
    float* cn  = hn + BH;                    // (Bz, H)

    // Workspace layout (all offsets 256B-aligned)
    char* ws = (char*)d_ws;
    size_t off = 0;
    float*  gx    = (float*)(ws + off);  off += (size_t)M1 * G * sizeof(float);   // 512 MB
    __bf16* xb    = (__bf16*)(ws + off); off += (size_t)M1 * I * sizeof(__bf16);  // 64 MB
    __bf16* wb_ih = (__bf16*)(ws + off); off += (size_t)G * I * sizeof(__bf16);   // 8 MB
    __bf16* wb_hh = (__bf16*)(ws + off); off += (size_t)G * H * sizeof(__bf16);   // 8 MB
    float*  bias  = (float*)(ws + off);  off += (size_t)G * sizeof(float);
    __bf16* hb    = (__bf16*)(ws + off); off += (size_t)BH * sizeof(__bf16);
    float*  c     = (float*)(ws + off);  off += (size_t)BH * sizeof(float);
    float*  gates = (float*)(ws + off);  off += (size_t)Bz * G * sizeof(float);
    (void)ws_size; (void)in_sizes; (void)n_in; (void)out_size;

    // --- Prep ---
    prep_weights<<<(G * I + 255) / 256, 256, 0, stream>>>(
        w_ih, w_hh, hh_mask, b_ih, b_hh, wb_ih, wb_hh, bias);
    conv_x<<<(M1 * I + 255) / 256, 256, 0, stream>>>(x, xb, M1 * I);
    init_state<<<(BH + 255) / 256, 256, 0, stream>>>(c, hb);

    // --- Phase 1: one big GEMM for all timesteps ---
    // waves = (M1/16) * (G/64) = 2048*64 = 131072 -> 16384 blocks of 8 waves
    gx_gemm<<<16384, 256, 0, stream>>>(xb, wb_ih, bias, gx);

    // --- Phase 2: sequential scan ---
    // step waves = (Bz/16) * (G/64) = 4*64 = 256 -> 32 blocks of 8 waves
    for (int t = 0; t < T; ++t) {
        step_gemm<<<32, 256, 0, stream>>>(hb, wb_hh, gx + (size_t)t * Bz * G, gates);
        cell_update<<<(BH + 255) / 256, 256, 0, stream>>>(
            gates, c, hb, out + (size_t)t * BH, hn, cn, t == T - 1 ? 1 : 0);
    }
}
